// DCGRUCell_36928128811052
// MI455X (gfx1250) — compile-verified, matrix operand order checked by reading, not run on hardware
//
#include <hip/hip_runtime.h>
#include <hip/hip_bf16.h>

typedef __attribute__((ext_vector_type(16))) _Float16 v16h;
typedef __attribute__((ext_vector_type(8)))  _Float16 v8h;
typedef __attribute__((ext_vector_type(8)))  float    v8f;
typedef __attribute__((ext_vector_type(4)))  unsigned int v4u;
typedef __attribute__((ext_vector_type(8)))  int      v8i;
typedef __attribute__((ext_vector_type(4)))  int      v4i;

#define N_NODES 3000
#define NP      3072                  // nodes padded to multiple of 128 (block tile)
#define N_FEAT  2
#define N_UNITS 64
#define BATCH   16
#define C_IN    (N_FEAT + N_UNITS)    // 66
#define CB      (C_IN * BATCH)        // 1056  (col = c*16 + b)
#define M_MATS  5
#define KPROJ   (C_IN * M_MATS)       // 330
#define KPROJ_P 352                   // padded to multiple of 32
#define ROWS    (BATCH * N_NODES)     // 48000

#if defined(__has_builtin)
#  if __has_builtin(__builtin_amdgcn_tensor_load_to_lds) && \
      __has_builtin(__builtin_amdgcn_s_wait_tensorcnt)
#    define HAVE_TDM 1
#  endif
#endif
#ifndef HAVE_TDM
#  define HAVE_TDM 0
#endif

// ---------------------------------------------------------------------------
// Generic f16 WMMA GEMM:  acc = A(MxK) * B(KxN), f32 accumulate.
// MODE 0: outH = (f16)(alpha*acc + beta*Z)   (Chebyshev combine, f16 out)
// MODE 1: outF = acc                         (raw f32 out for projection)
// Block tile 128x128x32, 256 threads = 8 waves, wave tile 32x64 (2x4 WMMAs).
// A tile (128x32 f16, row stride lda) is staged to LDS by the Tensor Data
// Mover: wave 0 issues one tensor_load_to_lds per k-step; D#.pad_enable
// inserts the 16B LDS row pad (pad_interval=3 -> every 64B, pad_amount=3 ->
// 16B) so the TDM writes directly into the As[128][40] layout.
// Requires: M % 128 == 0, K % 32 == 0, N % 8 == 0 (true for all call sites).
// ---------------------------------------------------------------------------
template<int MODE>
__global__ __launch_bounds__(256)
void gemm_wmma(const _Float16* __restrict__ A, int lda,
               const _Float16* __restrict__ B, int ldb,
               float alpha, float beta,
               const _Float16* __restrict__ Z, int ldz,
               _Float16* __restrict__ outH, float* __restrict__ outF, int ldc,
               int M, int N, int K)
{
    __shared__ _Float16 As[128][40];   // [m][k], +8 halves pad
    __shared__ _Float16 Bs[128][40];   // transposed: [n][k]

    const int t    = threadIdx.x;
    const int lane = t & 31;
    const int wave = t >> 5;
    const int wm   = wave & 3;         // 4 waves down M
    const int wn   = wave >> 2;        // 2 waves across N
    const int m0   = blockIdx.x * 128;
    const int n0   = blockIdx.y * 128;

    v8f acc[2][4];
#pragma unroll
    for (int i = 0; i < 2; ++i)
#pragma unroll
        for (int j = 0; j < 4; ++j)
            acc[i][j] = (v8f)0.0f;

#if HAVE_TDM
    // LDS aperture: low 32 bits of a flat shared pointer are the LDS offset.
    const unsigned int ldsA = (unsigned int)(uintptr_t)&As[0][0];
#endif

    for (int k0 = 0; k0 < K; k0 += 32) {
#if HAVE_TDM
        // ---- stage A tile 128x32 via Tensor Data Mover (wave 0 only) ----
        if (wave == 0) {
            unsigned long long ga =
                (unsigned long long)(uintptr_t)(A + (size_t)m0 * lda + k0);
            v4u g0;
            g0[0] = 1u;                                  // count=1, user D#
            g0[1] = ldsA;                                // lds_addr
            g0[2] = (unsigned int)ga;                    // global_addr[31:0]
            g0[3] = (unsigned int)((ga >> 32) & 0x1FFFFFFu) | (2u << 30); // [56:32]|type=2
            v8i g1;
            g1[0] = (1 << 16)      // data_size = 2 bytes
                  | (1 << 20)      // pad_enable
                  | (3 << 22)      // pad_interval: every 16 DWORDs (64B)
                  | (3 << 25);     // pad_amount: 4 DWORDs (16B)
            g1[1] = (32 & 0xFFFF) << 16;                 // tensor_dim0[15:0] = 32
            g1[2] = (128 & 0xFFFF) << 16;                // dim0 hi=0, tensor_dim1[15:0]=128
            g1[3] = (32 << 16);                          // dim1 hi=0, tile_dim0 = 32
            g1[4] = 128;                                 // tile_dim1 = 128, tile_dim2 = 0
            g1[5] = lda;                                 // tensor_dim0_stride[31:0]
            g1[6] = 0;                                   // stride hi / dim1_stride lo
            g1[7] = 0;
            v4i g2 = (v4i)0;                             // 2-D tensor: groups 2/3 unused
            v4i g3 = (v4i)0;
            v8i g4 = (v8i)0;                             // extra descriptor words (unused)
            __builtin_amdgcn_tensor_load_to_lds(g0, g1, g2, g3, g4, 0);
        }
#else
        // ---- fallback: stage A tile with vector loads ----
#pragma unroll
        for (int p = 0; p < 2; ++p) {
            int cid = p * 256 + t;
            int row = cid >> 2;
            int ch  = cid & 3;
            uint4 val = *(const uint4*)(A + (size_t)(m0 + row) * lda + k0 + ch * 8);
            *(uint4*)&As[row][ch * 8] = val;
        }
#endif
        // ---- stage B tile 32x128, stored transposed [n][k] ----
#pragma unroll
        for (int p = 0; p < 2; ++p) {
            int cid = p * 256 + t;
            int g   = cid & 15;        // column group of 8 (fast -> coalesced)
            int kk  = cid >> 4;        // k within tile
            int n   = n0 + g * 8;
            uint4 val = make_uint4(0u, 0u, 0u, 0u);
            if (n + 8 <= N)
                val = *(const uint4*)(B + (size_t)(k0 + kk) * ldb + n);
            union { uint4 u; _Float16 h[8]; } cv; cv.u = val;
#pragma unroll
            for (int j = 0; j < 8; ++j) Bs[g * 8 + j][kk] = cv.h[j];
        }
        if (k0 + 32 < K)
            __builtin_prefetch(B + (size_t)(k0 + 32 + (t >> 4)) * ldb + n0, 0, 1);
#if HAVE_TDM
        if (wave == 0)
            __builtin_amdgcn_s_wait_tensorcnt(0);        // A tile landed in LDS
#endif
        __syncthreads();

        // ---- build fragments per ISA 16-bit layout:
        // lane L: row/col = L&15 ; halves 0..7 = K[kb..kb+7], 8..15 = K[16+kb..],
        // kb = (L>>4)*8  -> two contiguous 16B LDS reads per fragment.
        const int kb = (lane >> 4) * 8;
        const int rc = lane & 15;
        v16h a[2], b[4];
#pragma unroll
        for (int i = 0; i < 2; ++i) {
            const _Float16* pr = &As[wm * 32 + i * 16 + rc][0];
            union { v16h v; v8h h[2]; } u;
            u.h[0] = *(const v8h*)(pr + kb);
            u.h[1] = *(const v8h*)(pr + 16 + kb);
            a[i] = u.v;
        }
#pragma unroll
        for (int j = 0; j < 4; ++j) {
            const _Float16* pr = &Bs[wn * 64 + j * 16 + rc][0];
            union { v16h v; v8h h[2]; } u;
            u.h[0] = *(const v8h*)(pr + kb);
            u.h[1] = *(const v8h*)(pr + 16 + kb);
            b[j] = u.v;
        }
#pragma unroll
        for (int i = 0; i < 2; ++i)
#pragma unroll
            for (int j = 0; j < 4; ++j)
                acc[i][j] = __builtin_amdgcn_wmma_f32_16x16x32_f16(
                    false, a[i], false, b[j], (short)0, acc[i][j], false, false);
        __syncthreads();
    }

    // ---- writeback: C/D layout: VGPR r -> M = r + 8*(lane>>4), N = lane&15 ----
    const int rbase = m0 + wm * 32 + ((lane >> 4) * 8);
    const int cbase = n0 + wn * 64 + (lane & 15);
#pragma unroll
    for (int i = 0; i < 2; ++i)
#pragma unroll
        for (int j = 0; j < 4; ++j) {
            int col = cbase + j * 16;
            if (col >= N) continue;
#pragma unroll
            for (int r = 0; r < 8; ++r) {
                int row = rbase + i * 16 + r;
                if (row >= M) continue;
                float v = acc[i][j][r];
                if (MODE == 0) {
                    v *= alpha;
                    if (Z) v += beta * (float)Z[(size_t)row * ldz + col];
                    outH[(size_t)row * ldc + col] = (_Float16)v;
                } else {
                    outF[(size_t)row * ldc + col] = v;
                }
            }
        }
}

// ---------------------------------------------------------------------------
// Support construction
// ---------------------------------------------------------------------------
__global__ void k_rowsum(const float* __restrict__ adj, float* __restrict__ rinv)
{
    __shared__ float red[256];
    int row = blockIdx.x;
    float s = 0.f;
    for (int j = threadIdx.x; j < N_NODES; j += 256)
        s += adj[(size_t)row * N_NODES + j];
    red[threadIdx.x] = s;
    __syncthreads();
    for (int off = 128; off > 0; off >>= 1) {
        if (threadIdx.x < off) red[threadIdx.x] += red[threadIdx.x + off];
        __syncthreads();
    }
    if (threadIdx.x == 0) {
        float d = red[0];
        rinv[row] = d > 0.f ? 1.f / d : 0.f;
    }
}

__global__ void k_colsum(const float* __restrict__ adj, float* __restrict__ cinv)
{
    int j = blockIdx.x * blockDim.x + threadIdx.x;
    if (j >= N_NODES) return;
    float s = 0.f;
    for (int i = 0; i < N_NODES; ++i) s += adj[(size_t)i * N_NODES + j];
    cinv[j] = s > 0.f ? 1.f / s : 0.f;
}

// S1[m][n] = adj[n][m] * rinv[n] (LDS-tiled transpose), S2[m][n] = adj[m][n]*cinv[n]
__global__ void k_build_S(const float* __restrict__ adj,
                          const float* __restrict__ rinv,
                          const float* __restrict__ cinv,
                          _Float16* __restrict__ S1, _Float16* __restrict__ S2)
{
    __shared__ float tile[32][33];
    int mB = blockIdx.x * 32, nB = blockIdx.y * 32;
    int tx = threadIdx.x, ty = threadIdx.y;
    {
        int n = nB + ty, m = mB + tx;
        tile[ty][tx] = (n < N_NODES && m < N_NODES) ? adj[(size_t)n * N_NODES + m] : 0.f;
    }
    __syncthreads();
    {
        int m = mB + ty, n = nB + tx;
        float rv = (n < N_NODES) ? rinv[n] : 0.f;
        S1[(size_t)m * NP + n] = (_Float16)(tile[tx][ty] * rv);
        float v = (m < N_NODES && n < N_NODES) ? adj[(size_t)m * N_NODES + n] * cinv[n] : 0.f;
        S2[(size_t)m * NP + n] = (_Float16)v;
    }
}

// X0[node][c*16+b] = concat(inputs, state); padded node rows -> 0
__global__ void k_build_X0(const float* __restrict__ inputs,
                           const float* __restrict__ state,
                           _Float16* __restrict__ X0)
{
    int gid = blockIdx.x * blockDim.x + threadIdx.x;
    if (gid >= NP * CB) return;
    int node = gid / CB, col = gid % CB;
    int c = col >> 4, b = col & 15;
    float v = 0.f;
    if (node < N_NODES)
        v = (c < N_FEAT) ? inputs[(size_t)b * (N_NODES * N_FEAT) + node * N_FEAT + c]
                         : state[(size_t)b * (N_NODES * N_UNITS) + node * N_UNITS + (c - N_FEAT)];
    X0[gid] = (_Float16)v;
}

// xs[b*3000+node][c*5+m] = mats[m][node][c*16+b], K padded to 352 with zeros
__global__ void k_repack(const _Float16* __restrict__ m0, const _Float16* __restrict__ m1,
                         const _Float16* __restrict__ m2, const _Float16* __restrict__ m3,
                         const _Float16* __restrict__ m4, _Float16* __restrict__ xs)
{
    int gid = blockIdx.x * blockDim.x + threadIdx.x;
    if (gid >= ROWS * KPROJ_P) return;
    int r = gid / KPROJ_P, k = gid % KPROJ_P;
    _Float16 v = (_Float16)0.f;
    if (k < KPROJ) {
        int c = k / M_MATS, mm = k % M_MATS;
        int b = r / N_NODES, node = r % N_NODES;
        const _Float16* mp = (mm == 0) ? m0 : (mm == 1) ? m1 : (mm == 2) ? m2
                           : (mm == 3) ? m3 : m4;
        v = mp[(size_t)node * CB + c * 16 + b];
    }
    xs[gid] = v;
}

__global__ void k_build_W(const float* __restrict__ Wg, const float* __restrict__ Wc,
                          _Float16* __restrict__ Whg, _Float16* __restrict__ Whc)
{
    int gid = blockIdx.x * blockDim.x + threadIdx.x;
    if (gid >= KPROJ_P * 128) return;
    int k = gid / 128, n = gid % 128;
    Whg[gid] = (_Float16)((k < KPROJ) ? Wg[(size_t)k * 128 + n] : 0.f);
    Whc[gid] = (_Float16)((k < KPROJ && n < N_UNITS) ? Wc[(size_t)k * N_UNITS + n] : 0.f);
}

// sigmoid gates; store u; overwrite state columns of X0 in-place with r*hx
__global__ void k_gate_epi(const float* __restrict__ Y, const float* __restrict__ bg,
                           const float* __restrict__ hx,
                           float* __restrict__ u_buf, _Float16* __restrict__ X0c)
{
    int idx = blockIdx.x * blockDim.x + threadIdx.x;
    if (idx >= ROWS * N_UNITS) return;
    int r = idx >> 6, unit = idx & 63;
    int b = r / N_NODES, node = r % N_NODES;
    float rr = Y[(size_t)r * 128 + unit] + bg[unit];
    float uu = Y[(size_t)r * 128 + 64 + unit] + bg[64 + unit];
    rr = 1.f / (1.f + __expf(-rr));
    uu = 1.f / (1.f + __expf(-uu));
    float h = hx[(size_t)b * (N_NODES * N_UNITS) + node * N_UNITS + unit];
    u_buf[idx] = uu;
    X0c[(size_t)node * CB + (unit + N_FEAT) * 16 + b] = (_Float16)(rr * h);
}

__global__ void k_final(const float* __restrict__ Y, const float* __restrict__ bc,
                        const float* __restrict__ u_buf, const float* __restrict__ hx,
                        float* __restrict__ out)
{
    int idx = blockIdx.x * blockDim.x + threadIdx.x;
    if (idx >= ROWS * N_UNITS) return;
    int r = idx >> 6, unit = idx & 63;
    int b = r / N_NODES, node = r % N_NODES;
    float c = tanhf(Y[(size_t)r * 128 + unit] + bc[unit]);
    float u = u_buf[idx];
    size_t hidx = (size_t)b * (N_NODES * N_UNITS) + node * N_UNITS + unit;
    out[hidx] = u * hx[hidx] + (1.f - u) * c;
}

// ---------------------------------------------------------------------------
extern "C" void kernel_launch(void* const* d_in, const int* in_sizes, int n_in,
                              void* d_out, int out_size, void* d_ws, size_t ws_size,
                              hipStream_t stream)
{
    const float* inputs = (const float*)d_in[0];
    const float* hx     = (const float*)d_in[1];
    const float* adj    = (const float*)d_in[2];
    const float* Wg     = (const float*)d_in[3];
    const float* bg     = (const float*)d_in[4];
    const float* Wc     = (const float*)d_in[5];
    const float* bc     = (const float*)d_in[6];
    float* out = (float*)d_out;

    char* p = (char*)d_ws;
    auto carve = [&](size_t bytes) -> char* {
        char* q = p;
        p += (bytes + 255) & ~(size_t)255;
        return q;
    };
    float*     rinv = (float*)carve((size_t)N_NODES * 4);
    float*     cinv = (float*)carve((size_t)N_NODES * 4);
    _Float16*  S1   = (_Float16*)carve((size_t)NP * NP * 2);
    _Float16*  S2   = (_Float16*)carve((size_t)NP * NP * 2);
    _Float16*  X0   = (_Float16*)carve((size_t)NP * CB * 2);
    _Float16*  X1a  = (_Float16*)carve((size_t)NP * CB * 2);
    _Float16*  X2a  = (_Float16*)carve((size_t)NP * CB * 2);
    _Float16*  X1b  = (_Float16*)carve((size_t)NP * CB * 2);
    _Float16*  X2b  = (_Float16*)carve((size_t)NP * CB * 2);
    _Float16*  xs   = (_Float16*)carve((size_t)ROWS * KPROJ_P * 2);
    _Float16*  Whg  = (_Float16*)carve((size_t)KPROJ_P * 128 * 2);
    _Float16*  Whc  = (_Float16*)carve((size_t)KPROJ_P * 128 * 2);
    float*     Y    = (float*)carve((size_t)ROWS * 128 * 4);
    float*     uB   = (float*)carve((size_t)ROWS * N_UNITS * 4);

    // supports (f16, padded to 3072; both fit in 192MB L2 afterwards)
    k_rowsum<<<N_NODES, 256, 0, stream>>>(adj, rinv);
    k_colsum<<<(N_NODES + 255) / 256, 256, 0, stream>>>(adj, cinv);
    k_build_S<<<dim3(NP / 32, NP / 32), dim3(32, 32), 0, stream>>>(adj, rinv, cinv, S1, S2);

    // gate pass
    k_build_X0<<<(NP * CB + 255) / 256, 256, 0, stream>>>(inputs, hx, X0);
    dim3 gDiff(NP / 128, (CB + 127) / 128);
    auto diffuse = [&](const _Float16* S, _Float16* x0, _Float16* x1, _Float16* x2) {
        gemm_wmma<0><<<gDiff, 256, 0, stream>>>(S, NP, x0, CB, 1.f, 0.f, nullptr, CB,
                                                x1, nullptr, CB, NP, CB, NP);
        gemm_wmma<0><<<gDiff, 256, 0, stream>>>(S, NP, x1, CB, 2.f, -1.f, x0, CB,
                                                x2, nullptr, CB, NP, CB, NP);
    };
    diffuse(S1, X0, X1a, X2a);
    diffuse(S2, X0, X1b, X2b);

    k_build_W<<<(KPROJ_P * 128 + 255) / 256, 256, 0, stream>>>(Wg, Wc, Whg, Whc);
    k_repack<<<(ROWS * KPROJ_P + 255) / 256, 256, 0, stream>>>(X0, X1a, X2a, X1b, X2b, xs);

    dim3 gProj(ROWS / 128, 1);
    gemm_wmma<1><<<gProj, 256, 0, stream>>>(xs, KPROJ_P, Whg, 128, 1.f, 0.f, nullptr, 128,
                                            nullptr, Y, 128, ROWS, 128, KPROJ_P);

    // candidate pass: overwrite only state columns of X0 with r*hx (inputs/pad stay)
    k_gate_epi<<<(ROWS * N_UNITS + 255) / 256, 256, 0, stream>>>(Y, bg, hx, uB, X0);
    diffuse(S1, X0, X1a, X2a);
    diffuse(S2, X0, X1b, X2b);
    k_repack<<<(ROWS * KPROJ_P + 255) / 256, 256, 0, stream>>>(X0, X1a, X2a, X1b, X2b, xs);
    gemm_wmma<1><<<gProj, 256, 0, stream>>>(xs, KPROJ_P, Whc, 128, 1.f, 0.f, nullptr, 128,
                                            nullptr, Y, 128, ROWS, 128, KPROJ_P);

    k_final<<<(ROWS * N_UNITS + 255) / 256, 256, 0, stream>>>(Y, bc, uB, hx, out);
}
